// Correlation_48344151884073
// MI455X (gfx1250) — matrix operand order repaired and checked by reading, not compiled, not run on hardware
//
#include <hip/hip_runtime.h>

// FlowNetC correlation on gfx1250 via V_WMMA_F32_16X16X32_BF16.
// Block owns (b, 16-wide w tile, 4 h-rows spaced by 2). Loops over the union
// of x2 rows r = h0-20+2*rr; each staged row feeds 4 statically-unrolled
// 16x16x256 GEMM tiles with B fragments held in registers. A fragments are
// preloaded per tile (16 ds loads in flight -> single dscnt wait before the
// 8-WMMA chain). Band extraction is branchless via a scratch Os slot.

#define BB 4
#define CC 256
#define HH 96
#define WW 96
#define ND 21            // displacements per axis
#define NP (ND*ND)
#define RR 4             // h rows per block, spaced by 2 (same parity)

typedef __attribute__((ext_vector_type(4)))  float        v4f;
typedef __attribute__((ext_vector_type(2)))  unsigned int v2u;
typedef __attribute__((ext_vector_type(2)))  __bf16       v2bf;
typedef __attribute__((ext_vector_type(8)))  __bf16       v8bf;
typedef __attribute__((ext_vector_type(16))) __bf16       v16bf;
typedef __attribute__((ext_vector_type(8)))  float        v8f;

__device__ __forceinline__ unsigned int pk2(float a, float b) {
    // native fptrunc f32->bf16 (RNE); lowers to packed cvt when available
    v2bf t;
    t[0] = (__bf16)a;
    t[1] = (__bf16)b;
    return __builtin_bit_cast(unsigned int, t);
}

__device__ __forceinline__ unsigned short bf1(float a) {
    return __builtin_bit_cast(unsigned short, (__bf16)a);
}

__device__ __forceinline__ v16bf ldfrag(const unsigned short* p0, const unsigned short* p1) {
    v8bf lo = *(const v8bf*)p0;   // ds_load_b128
    v8bf hi = *(const v8bf*)p1;   // ds_load_b128
    return __builtin_shufflevector(lo, hi, 0,1,2,3,4,5,6,7,8,9,10,11,12,13,14,15);
}

__global__ __launch_bounds__(128)
void corr_wmma_bf16(const float* __restrict__ x1,
                    const float* __restrict__ x2,
                    float* __restrict__ out) {
    // +8 bf16 row pad -> conflict-free ds_load_b128 fragment reads.
    __shared__ unsigned short As[RR][16][CC + 8];   // A panels: [j][w][c]
    __shared__ unsigned short Bs[64][CC + 8];       // B panel:  [q][c]
    __shared__ float          Os[RR][ND + 1][16];   // band staging (+1 scratch slot)

    const int tid  = threadIdx.x;
    const int lane = tid & 31;
    const int wave = tid >> 5;         // 0..3 -> N tile
    const int half = lane >> 4;        // 0/1
    const int m    = lane & 15;        // A row (M) / B column (N)
    const int w0   = blockIdx.x * 16;
    const int gy   = blockIdx.y;       // 0..23
    const int h0   = ((gy >> 1) * 8) + (gy & 1);   // rows h0 + {0,2,4,6}
    const int b    = blockIdx.z;

    // ---- stage RR A panels once: As[j][w][c] = bf16(x1[b, c, h0+2j, w0+w]) ----
    for (int i = tid; i < RR * 16 * CC; i += 128) {
        const int w = i & 15;
        const int c = (i >> 4) & (CC - 1);
        const int j = i >> 12;
        As[j][w][c] = bf1(x1[((b * CC + c) * HH + (h0 + 2 * j)) * WW + w0 + w]);
    }
    __syncthreads();

    // B staging: thread covers a 4-col quad x 4 channels per iter. Aligned
    // quads are fully inside [0,W) or fully in zero-pad (w0%16==0, pad=20).
    const int  qv    = (tid & 15) * 4;       // q quad base: 0..60
    const int  tcg   = tid >> 4;             // 0..7 channel group
    const int  col0  = w0 + qv - 20;
    const bool colok = (col0 >= 0) && (col0 < WW);

    const unsigned short* brow = &Bs[16 * wave + m][0];
    const int   q     = 16 * wave + m;       // x2p col - w0 for this lane
    const float scale = 1.0f / (float)CC;

    for (int rr = 0; rr < RR + 20; ++rr) {
        const int r   = h0 - 20 + 2 * rr;                 // x2 row (uniform)
        const int jlo = (rr - 20 > 0) ? (rr - 20) : 0;
        const int jhi = (rr < RR - 1) ? rr : (RR - 1);

        if (r >= 0 && r < HH) {
            // ---- stage B panel: Bs[q][c] = bf16(x2[b, c, r, w0+q-20]) or 0 ----
            #pragma unroll
            for (int it = 0; it < 8; ++it) {
                const int c4 = (tcg + 8 * it) * 4;
                v4f va = {}, vb = {}, vc = {}, vd = {};
                if (colok) {
                    const float* p0 = x2 + (((b * CC + c4) * HH) + r) * WW + col0;
                    va = *(const v4f*)(p0);
                    vb = *(const v4f*)(p0 + HH * WW);
                    vc = *(const v4f*)(p0 + 2 * HH * WW);
                    vd = *(const v4f*)(p0 + 3 * HH * WW);
                }
                #pragma unroll
                for (int jj = 0; jj < 4; ++jj) {
                    v2u pkd;
                    pkd.x = pk2(va[jj], vb[jj]);
                    pkd.y = pk2(vc[jj], vd[jj]);
                    *(v2u*)&Bs[qv + jj][c4] = pkd;    // ds_store_b64
                }
            }
            __syncthreads();

            // ---- hoist this wave's B fragments into registers (reused RR times) ----
            v16bf bfrag[8];
            #pragma unroll
            for (int kk = 0; kk < 8; ++kk) {
                const int k0 = kk * 32;
                bfrag[kk] = ldfrag(brow + k0 + 8 * half,
                                   brow + k0 + 16 + 8 * half);
            }

            // ---- 4 static 16x16x256 GEMM tiles (unserved j never stored) ----
            #pragma unroll
            for (int j = 0; j < RR; ++j) {
                const unsigned short* arow = &As[j][m][0];
                // preload all A fragments -> one dscnt wait, WMMAs back-to-back
                v16bf afrag[8];
                #pragma unroll
                for (int kk = 0; kk < 8; ++kk) {
                    const int k0 = kk * 32;
                    afrag[kk] = ldfrag(arow + k0 + 8 * half,
                                       arow + k0 + 16 + 8 * half);
                }
                v8f acc = {};
                #pragma unroll
                for (int kk = 0; kk < 8; ++kk) {
                    acc = __builtin_amdgcn_wmma_f32_16x16x32_bf16(
                              false, afrag[kk], false, bfrag[kk],
                              (short)0, acc, false, false);
                }
                // branchless band extraction: invalid (w,dx) -> scratch slot ND
                #pragma unroll
                for (int g = 0; g < 8; ++g) {
                    const int  M     = g + 8 * half;          // w - w0
                    const int  dx    = q - M;
                    const bool valid = ((unsigned)dx <= 40u) && !(dx & 1);
                    const int  slot  = valid ? (dx >> 1) : ND;
                    Os[j][slot][M] = acc[g];                  // ds_store_b32
                }
            }
            __syncthreads();

            // ---- coalesced stores: 21x16 floats per served (h, dy) pair ----
            for (int j = jlo; j <= jhi; ++j) {
                const int dyi = rr - j;
                const int h   = h0 + 2 * j;
                for (int i = tid; i < ND * 16; i += 128) {
                    const int dxi = i >> 4, w = i & 15;
                    out[((b * NP + dyi * ND + dxi) * HH + h) * WW + w0 + w] =
                        Os[j][dxi][w] * scale;
                }
            }
            __syncthreads();   // protect Bs/Os reuse next staged row
        } else {
            // x2 row in zero padding: outputs are exactly zero
            for (int j = jlo; j <= jhi; ++j) {
                const int dyi = rr - j;
                const int h   = h0 + 2 * j;
                for (int i = tid; i < ND * 16; i += 128) {
                    const int dxi = i >> 4, w = i & 15;
                    out[((b * NP + dyi * ND + dxi) * HH + h) * WW + w0 + w] = 0.0f;
                }
            }
        }
    }
}

extern "C" void kernel_launch(void* const* d_in, const int* in_sizes, int n_in,
                              void* d_out, int out_size, void* d_ws, size_t ws_size,
                              hipStream_t stream) {
    const float* x1 = (const float*)d_in[0];
    const float* x2 = (const float*)d_in[1];
    float* out = (float*)d_out;

    dim3 grid(WW / 16, 24, BB);   // (6, 24, 4) = 576 blocks, 4 waves each
    dim3 block(128);
    corr_wmma_bf16<<<grid, block, 0, stream>>>(x1, x2, out);
}
